// MLA_30459908063714
// MI455X (gfx1250) — matrix operand order repaired and checked by reading, not compiled, not run on hardware
//
#include <hip/hip_runtime.h>
#include <hip/hip_bf16.h>

typedef __attribute__((ext_vector_type(16))) __bf16 v16bf;
typedef __attribute__((ext_vector_type(8)))  float  v8f;

#define D_MODEL   2048
#define NUM_HEADS 16
#define D_H       128
#define D_LATENT  512
#define BB        2
#define SS        2048
#define MTOT      (BB*SS)   // 4096

static __device__ __forceinline__ void split_bf16(float f, __bf16& hi, __bf16& lo) {
    hi = (__bf16)f;
    lo = (__bf16)(f - (float)hi);
}

// ---------------------------------------------------------------------------
// Generic GEMM:  C[bz] = A @ B[bz] + bias[bz]
//   A: [M,K] fp32 row-major (shared across batch)
//   B: [K,N] fp32 row-major,  bias: [N]
//   Output: fp32 C, or bf16 Cbf when Cbf != nullptr (k/v projections).
// Split-bf16 (hi/lo) compensated WMMA: 3x v_wmma_f32_16x16x32_bf16 per MAC.
// Block tile 128x128, BK=32, 256 threads = 8 waves (4x2), wave tile 32x64.
// ---------------------------------------------------------------------------
#define GBM 128
#define GBN 128
#define GBK 32
#define APAD 36   // 32 + 4 bf16 pad -> 72B row stride (4B aligned)

__global__ __launch_bounds__(256) void gemm_bias_wmma(
    const float* __restrict__ A, const float* __restrict__ Bw,
    const float* __restrict__ bias, float* __restrict__ C,
    __bf16* __restrict__ Cbf,
    int M, int N, int K,
    long sBb, long sBiasb, long sCb)
{
    __shared__ __bf16 sAh[GBM][APAD];
    __shared__ __bf16 sAl[GBM][APAD];
    __shared__ __bf16 sBh[GBN][APAD];
    __shared__ __bf16 sBl[GBN][APAD];

    const int bz = blockIdx.z;
    Bw   += (long)bz * sBb;
    bias += (long)bz * sBiasb;
    if (Cbf) Cbf += (long)bz * sCb; else C += (long)bz * sCb;

    const int m0   = blockIdx.x * GBM;
    const int n0   = blockIdx.y * GBN;
    const int tid  = threadIdx.x;
    const int lane = tid & 31;
    const int wave = tid >> 5;
    const int mw   = (wave >> 1) * 32;   // wave row offset (0..96)
    const int nw   = (wave & 1) * 64;    // wave col offset (0 or 64)
    const int lm   = lane & 15;
    const int aoff = (lane < 16) ? 0 : 8;    // A-frag K offset per lane half
    const int boff = (lane < 16) ? 0 : 8;    // B-frag K offset per lane half

    v8f acc[2][4];
    #pragma unroll
    for (int i = 0; i < 2; i++)
        #pragma unroll
        for (int j = 0; j < 4; j++)
            #pragma unroll
            for (int r = 0; r < 8; r++) acc[i][j][r] = 0.0f;

    const int ac4 = tid & 7;    // A: col group of 4 floats (0..7)
    const int ar0 = tid >> 3;   // A: row 0..31 (stride 32)
    const int bc4 = tid & 31;   // B: col group of 4 floats (0..31)
    const int br0 = tid >> 5;   // B: k-row 0..7 (stride 8)

    for (int k0 = 0; k0 < K; k0 += GBK) {
        // ---- stage A tile 128x32 (fp32 -> bf16 hi/lo) ----
        #pragma unroll
        for (int i = 0; i < 4; i++) {
            const int r = ar0 + 32 * i;
            const float4 f4 = *reinterpret_cast<const float4*>(
                &A[(long)(m0 + r) * K + k0 + ac4 * 4]);
            const float fv[4] = {f4.x, f4.y, f4.z, f4.w};
            #pragma unroll
            for (int j = 0; j < 4; j++) {
                __bf16 h, l; split_bf16(fv[j], h, l);
                sAh[r][ac4 * 4 + j] = h;
                sAl[r][ac4 * 4 + j] = l;
            }
        }
        // ---- stage B tile 32x128 transposed -> LDS [n][k] ----
        #pragma unroll
        for (int i = 0; i < 4; i++) {
            const int r = br0 + 8 * i;
            const float4 f4 = *reinterpret_cast<const float4*>(
                &Bw[(long)(k0 + r) * N + n0 + bc4 * 4]);
            const float fv[4] = {f4.x, f4.y, f4.z, f4.w};
            #pragma unroll
            for (int j = 0; j < 4; j++) {
                __bf16 h, l; split_bf16(fv[j], h, l);
                sBh[bc4 * 4 + j][r] = h;
                sBl[bc4 * 4 + j][r] = l;
            }
        }
        // prefetch next K tile (global_prefetch_b8)
        if (k0 + GBK < K) {
            __builtin_prefetch(&A[(long)(m0 + ar0) * K + k0 + GBK + ac4 * 4], 0, 1);
            __builtin_prefetch(&Bw[(long)(k0 + GBK + br0) * N + n0 + bc4 * 4], 0, 1);
        }
        __syncthreads();

        // ---- fragments ----
        v16bf ah[2], al[2], bh[4], bl[4];
        #pragma unroll
        for (int mt = 0; mt < 2; mt++) {
            const uint32_t* ph = (const uint32_t*)&sAh[mw + mt * 16 + lm][0];
            const uint32_t* pl = (const uint32_t*)&sAl[mw + mt * 16 + lm][0];
            union { v16bf v; uint32_t u[8]; } fh, fl;
            #pragma unroll
            for (int i = 0; i < 4; i++) { fh.u[i]     = ph[(aoff >> 1) + i];
                                          fl.u[i]     = pl[(aoff >> 1) + i]; }
            #pragma unroll
            for (int i = 0; i < 4; i++) { fh.u[4 + i] = ph[((aoff + 16) >> 1) + i];
                                          fl.u[4 + i] = pl[((aoff + 16) >> 1) + i]; }
            ah[mt] = fh.v; al[mt] = fl.v;
        }
        #pragma unroll
        for (int nt = 0; nt < 4; nt++) {
            const uint32_t* ph = (const uint32_t*)&sBh[nw + nt * 16 + lm][0];
            const uint32_t* pl = (const uint32_t*)&sBl[nw + nt * 16 + lm][0];
            union { v16bf v; uint32_t u[8]; } fh, fl;
            #pragma unroll
            for (int i = 0; i < 8; i++) { fh.u[i] = ph[(boff >> 1) + i + (i & 4)];
                                          fl.u[i] = pl[(boff >> 1) + i + (i & 4)]; }
            bh[nt] = fh.v; bl[nt] = fl.v;
        }

        // ---- 3-term compensated WMMA (24 per k-step) ----
        #pragma unroll
        for (int mt = 0; mt < 2; mt++)
            #pragma unroll
            for (int nt = 0; nt < 4; nt++) {
                acc[mt][nt] = __builtin_amdgcn_wmma_f32_16x16x32_bf16(
                    false, ah[mt], false, bh[nt], (short)0, acc[mt][nt], false, false);
                acc[mt][nt] = __builtin_amdgcn_wmma_f32_16x16x32_bf16(
                    false, ah[mt], false, bl[nt], (short)0, acc[mt][nt], false, false);
                acc[mt][nt] = __builtin_amdgcn_wmma_f32_16x16x32_bf16(
                    false, al[mt], false, bh[nt], (short)0, acc[mt][nt], false, false);
            }
        __syncthreads();
    }

    // ---- epilogue: C/D layout VGPR r -> M=r (+8 for lanes 16..31), N=lane%16 ----
    const int halfsel = lane >> 4;
    #pragma unroll
    for (int mt = 0; mt < 2; mt++)
        #pragma unroll
        for (int nt = 0; nt < 4; nt++) {
            const int col = n0 + nw + nt * 16 + lm;
            const float bv = bias[col];
            if (Cbf) {
                #pragma unroll
                for (int r = 0; r < 8; r++) {
                    const int row = m0 + mw + mt * 16 + halfsel * 8 + r;
                    Cbf[(long)row * N + col] = (__bf16)(acc[mt][nt][r] + bv);
                }
            } else {
                #pragma unroll
                for (int r = 0; r < 8; r++) {
                    const int row = m0 + mw + mt * 16 + halfsel * 8 + r;
                    C[(long)row * N + col] = acc[mt][nt][r] + bv;
                }
            }
        }
}

// ---------------------------------------------------------------------------
// Flash attention per (b,h). Block = 4 waves x 16 q-rows = 64 q rows.
// K/V arrive pre-converted to bf16: k/v: [H][B*S][D_H] bf16.
//   qbuf: [B*S][D_MODEL] fp32 (head slice contiguous), ctx: [B*S][D_MODEL] fp32
// ---------------------------------------------------------------------------
#define KT 64

__global__ __launch_bounds__(128) void mla_attn_wmma(
    const float*  __restrict__ qbuf,
    const __bf16* __restrict__ kbuf,
    const __bf16* __restrict__ vbuf,
    float* __restrict__ ctx)
{
    __shared__ __bf16 sK[KT][D_H + 8];        // [key][d]   row 272B (16B aligned)
    __shared__ __bf16 sV[D_H][KT + 8];        // [d][key]   row 144B
    __shared__ __bf16 sP[4][16][KT + 4];      // per-wave P tile, row 136B

    const int qblk = blockIdx.x;              // 0..S/64-1
    const int bh   = blockIdx.y;              // 0..B*H-1
    const int b    = bh / NUM_HEADS;
    const int h    = bh % NUM_HEADS;

    const int tid  = threadIdx.x;
    const int wave = tid >> 5;
    const int lane = tid & 31;
    const int lm   = lane & 15;
    const int half = lane >> 4;
    const int aoff = half ? 8 : 0;
    const int boff = half ? 8 : 0;

    const float scale = 0.08838834764831845f;   // 1/sqrt(128)

    // ---- load q as A fragments (scaled), 16 rows per wave, K=d (4 chunks of 32)
    const int   sq = qblk * 64 + wave * 16 + lm;
    const float* gq = qbuf + (long)(b * SS + sq) * D_MODEL + h * D_H;
    v16bf qf[4];
    #pragma unroll
    for (int c = 0; c < 4; c++) {
        union { v16bf v; __bf16 e[16]; } f;
        #pragma unroll
        for (int i = 0; i < 8; i++) {
            f.e[i]     = (__bf16)(gq[c * 32 + aoff + i]      * scale);
            f.e[8 + i] = (__bf16)(gq[c * 32 + aoff + 16 + i] * scale);
        }
        qf[c] = f.v;
    }

    float mrow[8], lrow[8];
    v8f   oacc[8];
    #pragma unroll
    for (int r = 0; r < 8; r++) { mrow[r] = -3.0e38f; lrow[r] = 0.0f; }
    #pragma unroll
    for (int t = 0; t < 8; t++)
        #pragma unroll
        for (int r = 0; r < 8; r++) oacc[t][r] = 0.0f;

    const long kvbase = ((long)h * MTOT + (long)b * SS) * D_H;

    for (int j0 = 0; j0 < SS; j0 += KT) {
        // ---- stage K [key][d] (b128 copies) and V^T [d][key] tiles, bf16 ----
        for (int p = tid; p < KT * 16; p += 128) {
            const int row = p >> 4;       // key 0..63
            const int g   = p & 15;       // group of 8 bf16 along d
            const uint4 kk = *reinterpret_cast<const uint4*>(
                &kbuf[kvbase + (long)(j0 + row) * D_H + g * 8]);
            *reinterpret_cast<uint4*>(&sK[row][g * 8]) = kk;
            const uint4 vv = *reinterpret_cast<const uint4*>(
                &vbuf[kvbase + (long)(j0 + row) * D_H + g * 8]);
            const __bf16* ve = reinterpret_cast<const __bf16*>(&vv);
            #pragma unroll
            for (int jj = 0; jj < 8; jj++) sV[g * 8 + jj][row] = ve[jj];
        }
        __syncthreads();

        // ---- S = q @ K^T : 16 WMMAs ----
        v8f sc[4];
        #pragma unroll
        for (int nt = 0; nt < 4; nt++)
            #pragma unroll
            for (int r = 0; r < 8; r++) sc[nt][r] = 0.0f;

        #pragma unroll
        for (int c = 0; c < 4; c++) {
            v16bf kb[4];
            #pragma unroll
            for (int nt = 0; nt < 4; nt++) {
                const uint32_t* p = (const uint32_t*)&sK[nt * 16 + lm][0];
                union { v16bf v; uint32_t u[8]; } f;
                #pragma unroll
                for (int i = 0; i < 8; i++) f.u[i] = p[((c * 32 + boff) >> 1) + i + (i & 4)];
                kb[nt] = f.v;
            }
            #pragma unroll
            for (int nt = 0; nt < 4; nt++)
                sc[nt] = __builtin_amdgcn_wmma_f32_16x16x32_bf16(
                    false, qf[c], false, kb[nt], (short)0, sc[nt], false, false);
        }

        // ---- online softmax (row stats per VGPR-row; 16-lane half reduction) ----
        #pragma unroll
        for (int r = 0; r < 8; r++) {
            float mx = fmaxf(fmaxf(sc[0][r], sc[1][r]), fmaxf(sc[2][r], sc[3][r]));
            #pragma unroll
            for (int d = 8; d >= 1; d >>= 1) mx = fmaxf(mx, __shfl_xor(mx, d, 32));
            const float mnew = fmaxf(mrow[r], mx);
            const float corr = __expf(mrow[r] - mnew);
            float psum = 0.0f;
            #pragma unroll
            for (int nt = 0; nt < 4; nt++) {
                const float pvv = __expf(sc[nt][r] - mnew);
                psum += pvv;
                sP[wave][half * 8 + r][nt * 16 + lm] = (__bf16)pvv;
            }
            #pragma unroll
            for (int d = 8; d >= 1; d >>= 1) psum += __shfl_xor(psum, d, 32);
            lrow[r] = lrow[r] * corr + psum;
            mrow[r] = mnew;
            #pragma unroll
            for (int t = 0; t < 8; t++) oacc[t][r] *= corr;
        }

        // ---- O += P @ V : 16 WMMAs (K = 64 keys -> 2 chunks of 32) ----
        #pragma unroll
        for (int kc = 0; kc < 2; kc++) {
            const uint32_t* pp = (const uint32_t*)&sP[wave][lm][0];
            union { v16bf v; uint32_t u[8]; } fa;
            #pragma unroll
            for (int i = 0; i < 4; i++) fa.u[i]     = pp[((kc * 32 + aoff) >> 1) + i];
            #pragma unroll
            for (int i = 0; i < 4; i++) fa.u[4 + i] = pp[((kc * 32 + aoff + 16) >> 1) + i];
            #pragma unroll
            for (int t = 0; t < 8; t++) {
                const uint32_t* pv = (const uint32_t*)&sV[t * 16 + lm][0];
                union { v16bf v; uint32_t u[8]; } fb;
                #pragma unroll
                for (int i = 0; i < 8; i++) fb.u[i] = pv[((kc * 32 + boff) >> 1) + i + (i & 4)];
                oacc[t] = __builtin_amdgcn_wmma_f32_16x16x32_bf16(
                    false, fa.v, false, fb.v, (short)0, oacc[t], false, false);
            }
        }
        __syncthreads();
    }

    // ---- finalize: O / l -> ctx[b*S+s][h*128 + d] ----
    const long obase = (long)(b * SS + qblk * 64 + wave * 16) * D_MODEL + h * D_H;
    #pragma unroll
    for (int r = 0; r < 8; r++) {
        const float inv = 1.0f / lrow[r];
        const int   row = half * 8 + r;
        #pragma unroll
        for (int t = 0; t < 8; t++)
            ctx[obase + (long)row * D_MODEL + t * 16 + lm] = oacc[t][r] * inv;
    }
}

// ---------------------------------------------------------------------------
extern "C" void kernel_launch(void* const* d_in, const int* in_sizes, int n_in,
                              void* d_out, int out_size, void* d_ws, size_t ws_size,
                              hipStream_t stream) {
    (void)in_sizes; (void)n_in; (void)out_size; (void)ws_size;

    const float* x        = (const float*)d_in[0];
    const float* w_latent = (const float*)d_in[1];
    const float* b_latent = (const float*)d_in[2];
    const float* w_q      = (const float*)d_in[3];
    const float* b_q      = (const float*)d_in[4];
    const float* w_k      = (const float*)d_in[5];   // [H,512,128]
    const float* b_k      = (const float*)d_in[6];   // [H,128]
    const float* w_v      = (const float*)d_in[7];
    const float* b_v      = (const float*)d_in[8];
    const float* w_o      = (const float*)d_in[9];
    const float* b_o      = (const float*)d_in[10];

    float*  z   = (float*)d_ws;                          // [4096,512]  fp32
    float*  q   = z   + (size_t)MTOT * D_LATENT;         // [4096,2048] fp32
    float*  ctx = q   + (size_t)MTOT * D_MODEL;          // [4096,2048] fp32
    __bf16* kb  = (__bf16*)(ctx + (size_t)MTOT * D_MODEL);   // [H][4096][128] bf16
    __bf16* vb  = kb  + (size_t)MTOT * D_MODEL;              // [H][4096][128] bf16

    // z = x @ w_latent + b_latent
    gemm_bias_wmma<<<dim3(MTOT / GBM, D_LATENT / GBN, 1), 256, 0, stream>>>(
        x, w_latent, b_latent, z, nullptr, MTOT, D_LATENT, D_MODEL, 0, 0, 0);

    // q = x @ w_q + b_q   (kept [B*S, D_MODEL]; head slice contiguous)
    gemm_bias_wmma<<<dim3(MTOT / GBM, D_MODEL / GBN, 1), 256, 0, stream>>>(
        x, w_q, b_q, q, nullptr, MTOT, D_MODEL, D_MODEL, 0, 0, 0);

    // k[h] = z @ w_k[h] + b_k[h]  (batched over heads; bf16 output)
    gemm_bias_wmma<<<dim3(MTOT / GBM, 1, NUM_HEADS), 256, 0, stream>>>(
        z, w_k, b_k, nullptr, kb, MTOT, D_H, D_LATENT,
        (long)D_LATENT * D_H, (long)D_H, (long)MTOT * D_H);

    // v[h] = z @ w_v[h] + b_v[h]  (bf16 output)
    gemm_bias_wmma<<<dim3(MTOT / GBM, 1, NUM_HEADS), 256, 0, stream>>>(
        z, w_v, b_v, nullptr, vb, MTOT, D_H, D_LATENT,
        (long)D_LATENT * D_H, (long)D_H, (long)MTOT * D_H);

    // attention -> ctx [B*S, D_MODEL]
    mla_attn_wmma<<<dim3(SS / 64, BB * NUM_HEADS), 128, 0, stream>>>(q, kb, vb, ctx);

    // out = ctx @ w_o + b_o
    gemm_bias_wmma<<<dim3(MTOT / GBM, D_MODEL / GBN, 1), 256, 0, stream>>>(
        ctx, w_o, b_o, (float*)d_out, nullptr, MTOT, D_MODEL, D_MODEL, 0, 0, 0);
}
